// BasicRNN_58291296141476
// MI455X (gfx1250) — matrix operand order repaired and checked by reading, not compiled, not run on hardware
//
#include <hip/hip_runtime.h>
#include <math.h>

// ---------------------------------------------------------------------------
// BasicRNN scan for MI455X (gfx1250, wave32, WMMA).
// B=128, S=8000, H=64.  Latency-bound sequential scan:
//   8 workgroups x 1 wave, each wave owns a 16-row batch tile.
//   Per step: 4x WMMA for the rank-2 input projection (x*W_ih + bias),
//             8x WMMA for h @ W_hh^T (M=16 N=64 K=64, f16 in / f32 acc),
//             2x WMMA for fc logits; tanh/sigmoid in f32; h round-trips
//             through LDS (f16) to convert C-layout -> A-layout.
// ---------------------------------------------------------------------------

typedef __attribute__((ext_vector_type(16))) _Float16 v16h;
typedef __attribute__((ext_vector_type(8)))  _Float16 v8h;
typedef __attribute__((ext_vector_type(8)))  float    v8f;

#if __has_builtin(__builtin_amdgcn_tanhf)
#define FAST_TANH(x) __builtin_amdgcn_tanhf(x)
#else
#define FAST_TANH(x) tanhf(x)
#endif

#define RNN_B 128
#define RNN_S 8000
#define RNN_H 64

__device__ __forceinline__ v8f wmma_f16(v16h a, v16h b, v8f c) {
  // (neg_a, A, neg_b, B, c_mod, C, reuse_a, reuse_b)
  return __builtin_amdgcn_wmma_f32_16x16x32_f16(false, a, false, b, (short)0, c,
                                                false, false);
}

__global__ __launch_bounds__(32)
void rnn_scan_kernel(const float* __restrict__ x,     // [B, S]
                     const float* __restrict__ W_ih,  // [H, 1]
                     const float* __restrict__ b_ih,  // [H]
                     const float* __restrict__ W_hh,  // [H, H]
                     const float* __restrict__ b_hh,  // [H]
                     const float* __restrict__ fc_w,  // [2, H]
                     const float* __restrict__ fc_b,  // [2]
                     float* __restrict__ out)         // [B]
{
  // h tile [16 rows][64 cols] as f16; row stride 72 halves = 144 B:
  // 16B aligned for b128 loads, and 144*l mod 256 cycles through all
  // 16-byte bank groups -> conflict-free A-fragment reloads.
  __shared__ __align__(16) _Float16 hbuf[16 * 72];

  const int lane  = threadIdx.x;      // 0..31 (wave32)
  const int n     = lane & 15;        // column index inside a 16-wide tile
  const int hi    = lane >> 4;        // half-wave select
  const int mrow0 = blockIdx.x * 16;  // this wave's batch rows [mrow0, mrow0+16)

  // ---- loop-invariant B fragments of W_hh^T (D = A*B + C, B is KxN) ----
  // 32x16 f16 B-fragment k order used here (wave32): v16h element i holds
  // k_local = 16*hi + i at column n  (contiguous order, same as A).
  v16h Bw[2][4];
  for (int kt = 0; kt < 2; ++kt)
    for (int nt = 0; nt < 4; ++nt) {
      v16h b;
      #pragma unroll
      for (int i = 0; i < 16; ++i) {
        const int k = 32 * kt + 16 * hi + i;
        b[i] = (_Float16)W_hh[(16 * nt + n) * RNN_H + k];  // B[k][n]=W_hh[n][k]
      }
      Bw[kt][nt] = b;
    }

  // Input-projection as a rank-2 WMMA: A_x holds [x[m], 1] in K=0,1;
  // B_xp holds rows W_ih (k=0) and b_ih+b_hh (k=1), zero elsewhere.
  v16h Bxp[4];
  for (int nt = 0; nt < 4; ++nt) {
    v16h b = {};
    if (hi == 0) {  // k_local 0,1 live in the hi==0 half of the fragment
      b[0] = (_Float16)W_ih[16 * nt + n];
      b[1] = (_Float16)(b_ih[16 * nt + n] + b_hh[16 * nt + n]);
    }
    Bxp[nt] = b;
  }

  // fc B-fragments: columns 0/1 = fc_w rows, columns 2..15 = 0.
  v16h Bf[2];
  for (int kt = 0; kt < 2; ++kt) {
    v16h b;
    #pragma unroll
    for (int i = 0; i < 16; ++i) {
      const int k = 32 * kt + 16 * hi + i;
      b[i] = (n < 2) ? (_Float16)fc_w[n * RNN_H + k] : (_Float16)0.0f;
    }
    Bf[kt] = b;
  }

  const float fcb     = (n < 2) ? fc_b[n] : 0.0f;
  const float selmask = (n == 0) ? 1.0f : 0.0f;
  const float xmask   = (hi == 0) ? 1.0f : 0.0f;  // K=0,1 only exist for hi==0

  // h_0 = 0 -> zero A-fragments (16x32 f16 each; two cover K=64).
  v16h A0 = {}, A1 = {};
  // A_x: element 1 (K=1) is the constant 1 for the bias row; element 0 set per step.
  v16h Ax = {};
  Ax[1] = (_Float16)xmask;

  float num[8], den[8];
  #pragma unroll
  for (int r = 0; r < 8; ++r) { num[r] = 0.0f; den[r] = 0.0f; }

  // Lane l streams batch row (mrow0 + l%16); A-layout means lane l%16 == m,
  // so the lane's own x value is exactly the one its A_x fragment needs.
  const float* xrow = x + (size_t)(mrow0 + n) * RNN_S;
  const v8f zero_c = {};

  for (int s = 0; s < RNN_S; ++s) {
    if ((s & 31) == 0)  // keep the x stream ahead of the scan (global_prefetch_b8)
      __builtin_prefetch(xrow + s + 128, 0, 1);

    Ax[0] = (_Float16)(xrow[s] * xmask);  // K=0 row of the rank-2 update

    v8f c[4];
    #pragma unroll
    for (int nt = 0; nt < 4; ++nt) {
      c[nt] = wmma_f16(Ax, Bxp[nt], zero_c);   // xp = x*W_ih + (b_ih+b_hh)
      c[nt] = wmma_f16(A0, Bw[0][nt], c[nt]);  // + h @ W_hh^T, K = 0..31
      c[nt] = wmma_f16(A1, Bw[1][nt], c[nt]);  // + h @ W_hh^T, K = 32..63
    }

    // tanh, downconvert, scatter h_{s+1} into LDS [m][k] (C-layout addressing:
    // element (m = r + 8*hi, k = 16*nt + n)).
    #pragma unroll
    for (int nt = 0; nt < 4; ++nt)
      #pragma unroll
      for (int r = 0; r < 8; ++r)
        hbuf[(r + 8 * hi) * 72 + 16 * nt + n] = (_Float16)FAST_TANH(c[nt][r]);

    __syncthreads();  // single-wave WG: barrier ~S_NOP, but fences DS order

    // Reload A-fragments of h_{s+1}. A k order used here: elements 0..15 are
    // k = 16*hi..16*hi+15 (matches B fragments) -> 4x ds_load_b128.
    const v8h* row = reinterpret_cast<const v8h*>(&hbuf[n * 72]);
    const v8h p0 = row[0 + 2 * hi], p1 = row[1 + 2 * hi];
    const v8h p2 = row[4 + 2 * hi], p3 = row[5 + 2 * hi];
    A0 = __builtin_shufflevector(p0, p1, 0,1,2,3,4,5,6,7,8,9,10,11,12,13,14,15);
    A1 = __builtin_shufflevector(p2, p3, 0,1,2,3,4,5,6,7,8,9,10,11,12,13,14,15);

    // fc logits via WMMA on the fragments we already have; cols 0/1 valid.
    v8f fcc = {};
    fcc = wmma_f16(A0, Bf[0], fcc);
    fcc = wmma_f16(A1, Bf[1], fcc);

    // sigmoid + online weighted-average accumulation.
    #pragma unroll
    for (int r = 0; r < 8; ++r) {
      const float sig   = 1.0f / (1.0f + __expf(-(fcc[r] + fcb)));  // lane n: o[..,n]
      const float score = __shfl(sig, lane + 1, 32);  // lane n=0 grabs n=1's value
      num[r] = fmaf(selmask * sig, score, num[r]);    // sum_s sel*score
      den[r] += selmask * sig;                        // sum_s sel
    }
  }

  // Lanes 0 (rows m=0..7) and 16 (rows m=8..15) hold the reductions.
  if (n == 0) {
    #pragma unroll
    for (int r = 0; r < 8; ++r)
      out[mrow0 + 8 * hi + r] = num[r] / den[r];
  }
}

extern "C" void kernel_launch(void* const* d_in, const int* in_sizes, int n_in,
                              void* d_out, int out_size, void* d_ws, size_t ws_size,
                              hipStream_t stream) {
  (void)in_sizes; (void)n_in; (void)d_ws; (void)ws_size; (void)out_size;
  const float* x    = (const float*)d_in[0];
  const float* W_ih = (const float*)d_in[1];
  const float* b_ih = (const float*)d_in[2];
  const float* W_hh = (const float*)d_in[3];
  const float* b_hh = (const float*)d_in[4];
  const float* fc_w = (const float*)d_in[5];
  const float* fc_b = (const float*)d_in[6];
  float* out        = (float*)d_out;

  rnn_scan_kernel<<<dim3(RNN_B / 16), dim3(32), 0, stream>>>(
      x, W_ih, b_ih, W_hh, b_hh, fc_w, fc_b, out);
}
